// GodheadTransformer_35656818492145
// MI455X (gfx1250) — matrix operand order repaired and checked by reading, not compiled, not run on hardware
//
#include <hip/hip_runtime.h>

#define USE_ASYNC_LDS 1

// ---------------------------------------------------------------------------
// Types for CDNA5 WMMA (wave32, v_wmma_f32_16x16x32_bf16)
// ---------------------------------------------------------------------------
typedef __attribute__((ext_vector_type(16))) __bf16 bf16x16;
typedef __attribute__((ext_vector_type(8)))  __bf16 bf16x8;
typedef __attribute__((ext_vector_type(8)))  float  f32x8;

static constexpr int Lc  = 8;
static constexpr int Cc  = 384;
static constexpr int Hh  = 8;
static constexpr int HDc = 48;
static constexpr int Ee  = 4;
static constexpr int TDc = 256;
static constexpr int Vv  = 256;
static constexpr int Mm  = 56;     // memory slots
static constexpr int WINc= 64;     // causal attention window
static constexpr int Bb  = 64;
static constexpr int Tt  = 256;
static constexpr int Rr  = Bb * Tt;      // 16384 rows
static constexpr int C3  = 3 * Cc;       // 1152
static constexpr int F4  = 4 * Cc;       // 1536
#define ATT_LD 168                       // 168*2B = 336B, 16B aligned rows

__device__ inline f32x8 wmma_bf16(bf16x16 a, bf16x16 b, f32x8 c) {
  return __builtin_amdgcn_wmma_f32_16x16x32_bf16(false, a, false, b, (short)0, c,
                                                 false, false);
}

// A fragment 16x32 (MxK), row-major LDS tile with leading dim ld.
__device__ inline bf16x16 frag_a(const __bf16* lds, int row_base, int k0, int ld) {
  int lane = threadIdx.x & 31;
  const __bf16* p = lds + (row_base + (lane & 15)) * ld + k0 + ((lane >> 4) << 3);
  bf16x8 lo = *(const bf16x8*)p;
  bf16x8 hi = *(const bf16x8*)(p + 16);
  bf16x16 r;
#pragma unroll
  for (int i = 0; i < 8; ++i) { r[i] = lo[i]; r[i + 8] = hi[i]; }
  return r;
}

// B fragment 32x16 (KxN) read from an LDS tile stored TRANSPOSED: [col][k].
__device__ inline bf16x16 frag_b(const __bf16* ldsT, int col_base, int k0, int ld) {
  int lane = threadIdx.x & 31;
  const __bf16* p = ldsT + (col_base + (lane & 15)) * ld + k0 + ((lane >> 4) << 3);
  bf16x8 lo = *(const bf16x8*)p;
  bf16x8 hi = *(const bf16x8*)(p + 16);
  bf16x16 r;
#pragma unroll
  for (int i = 0; i < 8; ++i) { r[i] = lo[i]; r[i + 8] = hi[i]; }
  return r;
}

__device__ inline f32x8 zero8() {
  f32x8 z;
#pragma unroll
  for (int i = 0; i < 8; ++i) z[i] = 0.0f;
  return z;
}

#if USE_ASYNC_LDS
// Per-lane 16B async copy global -> LDS (tracked by ASYNCcnt).
__device__ inline void async_g2l_b128(const __bf16* lds_dst, const __bf16* gsrc) {
  unsigned lds_off = (unsigned)(unsigned long long)lds_dst;  // addr[31:0] = LDS offset
  unsigned long long ga = (unsigned long long)gsrc;
  asm volatile("global_load_async_to_lds_b128 %0, %1, off"
               :: "v"(lds_off), "v"(ga) : "memory");
}
__device__ inline void wait_async_le0() {
  asm volatile("s_wait_asynccnt 0" ::: "memory");
}
// Allow the newest 3 async ops (next chunk) to remain in flight; in-order
// completion guarantees the previous chunk's 3 ops have landed in LDS.
__device__ inline void wait_async_le3() {
  asm volatile("s_wait_asynccnt 3" ::: "memory");
}
#endif

// Stage one K-chunk (A: 16B, B^T: 32B per thread) into an LDS buffer.
__device__ inline void stage_chunk(const __bf16* asrc, const __bf16* bsrc,
                                   __bf16* adst, __bf16* bdst) {
#if USE_ASYNC_LDS
  async_g2l_b128(adst, asrc);
  async_g2l_b128(bdst, bsrc);
  async_g2l_b128(bdst + 8, bsrc + 8);
#else
  *(bf16x8*)adst = *(const bf16x8*)asrc;
  *(bf16x8*)bdst = *(const bf16x8*)bsrc;
  *(bf16x8*)(bdst + 8) = *(const bf16x8*)(bsrc + 8);
#endif
}

// ---------------------------------------------------------------------------
// Elementwise helpers
// ---------------------------------------------------------------------------
__global__ void k_f2bf(const float* __restrict__ in, __bf16* __restrict__ out, long n) {
  long i = (long)blockIdx.x * 256 + threadIdx.x;
  if (i < n) out[i] = (__bf16)in[i];
}

// Batched convert + transpose: out[mat][n*K+k] = in[mat][k*N+n]  (weights -> W^T)
__global__ void k_f2bfT(const float* __restrict__ in, __bf16* __restrict__ out,
                        int K, int N) {
  long mat = blockIdx.y;
  long i = (long)blockIdx.x * 256 + threadIdx.x;
  long total = (long)K * N;
  if (i >= total) return;
  long k = i / N, n = i - k * N;
  out[mat * total + n * K + k] = (__bf16)in[mat * total + i];
}

__global__ void k_zero_int(int* p, int n) {
  int i = blockIdx.x * blockDim.x + threadIdx.x;
  if (i < n) p[i] = 0;
}

// x[row][c] = tok_emb[tokens[row]][c] + pos_emb[row%T][c]
__global__ __launch_bounds__(128) void k_embed(const int* __restrict__ tokens,
                                               const float* __restrict__ tok_emb,
                                               const float* __restrict__ pos_emb,
                                               float* __restrict__ x) {
  int row = blockIdx.x;
  int tok = tokens[row];
  int t = row & (Tt - 1);
  for (int c = threadIdx.x; c < Cc; c += 128)
    x[(size_t)row * Cc + c] = tok_emb[(size_t)tok * Cc + c] + pos_emb[(size_t)t * Cc + c];
}

// ---------------------------------------------------------------------------
// LayerNorm over C=384, one block (128 threads) per row, bf16 output
// ---------------------------------------------------------------------------
__global__ __launch_bounds__(128) void k_layernorm(const float* __restrict__ x,
                                                   const float* __restrict__ w,
                                                   const float* __restrict__ b,
                                                   __bf16* __restrict__ out) {
  __shared__ float red[4];
  int row = blockIdx.x, tid = threadIdx.x;
  const float* xr = x + (size_t)row * Cc;
  float v0 = xr[tid], v1 = xr[tid + 128], v2 = xr[tid + 256];
  float s = v0 + v1 + v2;
#pragma unroll
  for (int o = 16; o; o >>= 1) s += __shfl_xor(s, o, 32);
  if ((tid & 31) == 0) red[tid >> 5] = s;
  __syncthreads();
  float mu = (red[0] + red[1] + red[2] + red[3]) * (1.0f / Cc);
  __syncthreads();
  float d0 = v0 - mu, d1 = v1 - mu, d2 = v2 - mu;
  float sq = d0 * d0 + d1 * d1 + d2 * d2;
#pragma unroll
  for (int o = 16; o; o >>= 1) sq += __shfl_xor(sq, o, 32);
  if ((tid & 31) == 0) red[tid >> 5] = sq;
  __syncthreads();
  float var = (red[0] + red[1] + red[2] + red[3]) * (1.0f / Cc);
  float rstd = rsqrtf(var + 1e-5f);
  __bf16* orow = out + (size_t)row * Cc;
  orow[tid]       = (__bf16)(d0 * rstd * w[tid]       + b[tid]);
  orow[tid + 128] = (__bf16)(d1 * rstd * w[tid + 128] + b[tid + 128]);
  orow[tid + 256] = (__bf16)(d2 * rstd * w[tid + 256] + b[tid + 256]);
}

// ---------------------------------------------------------------------------
// MoE gate: logits (Cx4 dot), softmax(4), top-2, normalized weights + lists
// ---------------------------------------------------------------------------
__global__ __launch_bounds__(256) void k_gate(const __bf16* __restrict__ h2,
                                              const float* __restrict__ gw,
                                              const float* __restrict__ gb,
                                              float* __restrict__ wout,
                                              int* __restrict__ counts,
                                              int* __restrict__ lists) {
  int wv = threadIdx.x >> 5, lane = threadIdx.x & 31;
  int row = blockIdx.x * 8 + wv;
  float acc0 = 0, acc1 = 0, acc2 = 0, acc3 = 0;
  for (int c = lane; c < Cc; c += 32) {
    float hv = (float)h2[(size_t)row * Cc + c];
    const float* g = gw + c * Ee;
    acc0 += hv * g[0]; acc1 += hv * g[1]; acc2 += hv * g[2]; acc3 += hv * g[3];
  }
#pragma unroll
  for (int o = 16; o; o >>= 1) {
    acc0 += __shfl_xor(acc0, o, 32); acc1 += __shfl_xor(acc1, o, 32);
    acc2 += __shfl_xor(acc2, o, 32); acc3 += __shfl_xor(acc3, o, 32);
  }
  if (lane == 0) {
    float p[4] = {acc0 + gb[0], acc1 + gb[1], acc2 + gb[2], acc3 + gb[3]};
    float m = fmaxf(fmaxf(p[0], p[1]), fmaxf(p[2], p[3]));
    float sum = 0;
#pragma unroll
    for (int e = 0; e < 4; ++e) { p[e] = expf(p[e] - m); sum += p[e]; }
    float inv = 1.0f / sum;
#pragma unroll
    for (int e = 0; e < 4; ++e) p[e] *= inv;
    int i1 = 0;
#pragma unroll
    for (int e = 1; e < 4; ++e) if (p[e] > p[i1]) i1 = e;
    int i2 = (i1 == 0) ? 1 : 0;
#pragma unroll
    for (int e = 0; e < 4; ++e) if (e != i1 && p[e] > p[i2]) i2 = e;
    float denom = p[i1] + p[i2] + 1e-9f;
#pragma unroll
    for (int e = 0; e < 4; ++e)
      wout[(size_t)row * Ee + e] = (e == i1 || e == i2) ? p[e] / denom : 0.0f;
    int p1 = atomicAdd(&counts[i1], 1); lists[i1 * Rr + p1] = row;
    int p2 = atomicAdd(&counts[i2], 1); lists[i2 * Rr + p2] = row;
  }
}

// ---------------------------------------------------------------------------
// Generic WMMA GEMM with PRE-TRANSPOSED weights (Bt = W^T, N x K row-major):
//   out(MxN) = ACT(A(MxK) @ W + bias) (+ residual / MoE scatter)
// Block: 256 threads = 8 waves; tile 64(M) x 128(N); K chunks of 32.
// Double-buffered LDS staging via GLOBAL_LOAD_ASYNC_TO_LDS_B128: chunk i+1's
// DMA overlaps chunk i's WMMAs (s_wait_asynccnt 3 keeps 3 newest in flight).
// ACT: 0=none 1=tanh 2=gelu ; RESMODE: 0 store, 1 +=, 2 scatter +=w*val
// ---------------------------------------------------------------------------
template <int ACT, int RESMODE, int AGATHER>
__global__ __launch_bounds__(256) void k_gemm(
    const __bf16* __restrict__ A, int lda,
    const __bf16* __restrict__ Bt, int N, int K, int M,
    const float* __restrict__ bias,
    float* __restrict__ outF, __bf16* __restrict__ outB,
    const int* __restrict__ gidx, const int* __restrict__ gcount,
    const float* __restrict__ wts, int expert) {
  int effM = gcount ? gcount[0] : M;
  int m0 = blockIdx.y * 64;
  if (m0 >= effM) return;
  int n0 = blockIdx.x * 128;
  int tid = threadIdx.x, lane = tid & 31, w = tid >> 5;
  int wm = w >> 2, wn = w & 3;   // 2x4 wave grid, each wave: 32x32 C tile

  __shared__ __align__(16) __bf16 As[2][64 * 32];
  __shared__ __align__(16) __bf16 Bts[2][128 * 40];   // [n][k], k-pitch 40

  f32x8 acc[2][2];
  acc[0][0] = zero8(); acc[0][1] = zero8(); acc[1][0] = zero8(); acc[1][1] = zero8();

  // staging roles
  int ar = tid >> 2, aseg = (tid & 3) << 3;        // A: row(0..63), 8-elem segment
  int bn = tid >> 1, bseg = (tid & 1) << 4;        // B^T: row(0..127), 16-elem segment
  int agrow = m0 + ar;
  int arow = agrow;
  if (AGATHER) arow = (agrow < effM) ? gidx[agrow] : 0;

  const __bf16* aBase = A + (size_t)arow * lda + aseg;
  const __bf16* bBase = Bt + (size_t)(n0 + bn) * K + bseg;
  __bf16* aDst[2] = {&As[0][ar * 32 + aseg], &As[1][ar * 32 + aseg]};
  __bf16* bDst[2] = {&Bts[0][bn * 40 + bseg], &Bts[1][bn * 40 + bseg]};

  int nk = K >> 5;
  stage_chunk(aBase, bBase, aDst[0], bDst[0]);
  for (int i = 0; i < nk; ++i) {
    int cur = i & 1;
    if (i + 1 < nk) {
      int k1 = (i + 1) << 5;
      __builtin_prefetch(aBase + k1 + 32, 0, 1);    // global_prefetch_b8 (chunk i+2)
      stage_chunk(aBase + k1, bBase + k1, aDst[cur ^ 1], bDst[cur ^ 1]);
#if USE_ASYNC_LDS
      wait_async_le3();
#endif
    } else {
#if USE_ASYNC_LDS
      wait_async_le0();
#endif
    }
    __syncthreads();
    bf16x16 a0 = frag_a(As[cur], wm * 32,      0, 32);
    bf16x16 a1 = frag_a(As[cur], wm * 32 + 16, 0, 32);
    bf16x16 f0 = frag_b(Bts[cur], wn * 32,      0, 40);
    bf16x16 f1 = frag_b(Bts[cur], wn * 32 + 16, 0, 40);
    acc[0][0] = wmma_bf16(a0, f0, acc[0][0]);
    acc[0][1] = wmma_bf16(a0, f1, acc[0][1]);
    acc[1][0] = wmma_bf16(a1, f0, acc[1][0]);
    acc[1][1] = wmma_bf16(a1, f1, acc[1][1]);
    __syncthreads();
  }

#pragma unroll
  for (int i = 0; i < 2; ++i)
#pragma unroll
    for (int j = 0; j < 2; ++j) {
      int tm = m0 + wm * 32 + i * 16;
      int tn = n0 + wn * 32 + j * 16;
      int col = tn + (lane & 15);
      int rbase = tm + ((lane >> 4) << 3);
      float bv = bias[col];
#pragma unroll
      for (int r = 0; r < 8; ++r) {
        int row = rbase + r;
        if (row >= effM) continue;
        float v = acc[i][j][r] + bv;
        if (ACT == 1) v = tanhf(v);
        else if (ACT == 2) v = 0.5f * v * (1.0f + erff(v * 0.70710678118f));
        if (RESMODE == 2) {
          int orow = gidx[row];
          float s = wts[(size_t)orow * Ee + expert];
          outF[(size_t)orow * N + col] += s * v;
          continue;
        }
        if (RESMODE == 1) {
          float o = outF[(size_t)row * N + col] + v;
          outF[(size_t)row * N + col] = o;
          v = o;
        } else if (outF) {
          outF[(size_t)row * N + col] = v;
        }
        if (outB) outB[(size_t)row * N + col] = (__bf16)v;
      }
    }
}

// ---------------------------------------------------------------------------
// Fused attention for one (b, h, 32-query tile).
// Key axis: slots 0..63 = mem (56 valid), slots 64..159 = window [q0-64, q0+31].
// ---------------------------------------------------------------------------
__global__ __launch_bounds__(128) void k_attn(const __bf16* __restrict__ qkv,
                                              const __bf16* __restrict__ memb,
                                              const float* __restrict__ gate,
                                              __bf16* __restrict__ yg) {
  int q0 = blockIdx.x * 32;
  int hh = blockIdx.y;
  int bb = blockIdx.z;
  int tid = threadIdx.x, lane = tid & 31, wv = tid >> 5;

  __shared__ __align__(16) __bf16 qs[32 * 64];
  __shared__ __align__(16) __bf16 ks[160 * 64];
  __shared__ __align__(16) __bf16 sc[32 * ATT_LD];
  __shared__ __align__(16) __bf16 vts[48 * 160];

  size_t rowb = (size_t)bb * Tt;
  int hoff = hh * HDc;
  int kstart = q0 - WINc;

  for (int i = tid; i < 32 * 64; i += 128) {
    int r = i >> 6, d = i & 63;
    float v = (d < HDc) ? (float)qkv[(rowb + q0 + r) * C3 + hoff + d] : 0.0f;
    qs[i] = (__bf16)v;
  }
  for (int i = tid; i < 160 * 64; i += 128) {
    int s = i >> 6, d = i & 63;
    float kv = 0.0f, vv = 0.0f;
    if (d < HDc) {
      if (s < 64) {
        if (s < Mm) { kv = (float)memb[s * Cc + hoff + d]; vv = kv; }
      } else {
        int kk = kstart + s - 64;
        if (kk >= 0) {
          size_t base = (rowb + kk) * C3 + hoff + d;
          kv = (float)qkv[base + Cc];
          vv = (float)qkv[base + 2 * Cc];
        }
      }
    }
    ks[s * 64 + d] = (__bf16)kv;
    if (d < HDc) vts[d * 160 + s] = (__bf16)vv;
  }
  __syncthreads();

  // Phase 1: scores = q @ k^T (2 x 10 tiles over 4 waves), scaled bf16 in LDS
  const float SCALE = 0.14433756729740643f;  // 48^-0.5
  for (int job = wv; job < 20; job += 4) {
    int mt = job & 1, nt = job >> 1;
    f32x8 acc = zero8();
#pragma unroll
    for (int k0 = 0; k0 < 64; k0 += 32) {
      bf16x16 a = frag_a(qs, mt * 16, k0, 64);
      bf16x16 b = frag_b(ks, nt * 16, k0, 64);
      acc = wmma_bf16(a, b, acc);
    }
    int col = nt * 16 + (lane & 15);
    int rb2 = mt * 16 + ((lane >> 4) << 3);
#pragma unroll
    for (int r = 0; r < 8; ++r)
      sc[(rb2 + r) * ATT_LD + col] = (__bf16)(acc[r] * SCALE);
  }
  __syncthreads();

  // Phase 2: masked softmax; wave wv owns rows [wv*8, wv*8+8)
  for (int rr = 0; rr < 8; ++rr) {
    int row = wv * 8 + rr;
    int qi = q0 + row;
    float v[5];
#pragma unroll
    for (int i = 0; i < 5; ++i) {
      int c = lane + (i << 5);
      bool valid;
      if (c < 64) valid = (c < Mm);
      else { int kk = kstart + c - 64; valid = (kk >= 0) && (kk <= qi) && (qi - kk <= WINc); }
      v[i] = valid ? (float)sc[row * ATT_LD + c] : -3.0e38f;
    }
    float m = v[0];
#pragma unroll
    for (int i = 1; i < 5; ++i) m = fmaxf(m, v[i]);
#pragma unroll
    for (int o = 16; o; o >>= 1) m = fmaxf(m, __shfl_xor(m, o, 32));
    float e[5], s = 0.0f;
#pragma unroll
    for (int i = 0; i < 5; ++i) { e[i] = (v[i] > -1.0e38f) ? expf(v[i] - m) : 0.0f; s += e[i]; }
#pragma unroll
    for (int o = 16; o; o >>= 1) s += __shfl_xor(s, o, 32);
    float inv = 1.0f / s;
#pragma unroll
    for (int i = 0; i < 5; ++i)
      sc[row * ATT_LD + lane + (i << 5)] = (__bf16)(e[i] * inv);
  }
  __syncthreads();

  // Phase 3: y = att @ v ; fused attn_gate
  for (int job = wv; job < 6; job += 4) {
    int mt = job & 1, nt = job >> 1;
    f32x8 acc = zero8();
#pragma unroll
    for (int k0 = 0; k0 < 160; k0 += 32) {
      bf16x16 a = frag_a(sc, mt * 16, k0, ATT_LD);
      bf16x16 b = frag_b(vts, nt * 16, k0, 160);
      acc = wmma_bf16(a, b, acc);
    }
    int col = nt * 16 + (lane & 15);
    int rb2 = mt * 16 + ((lane >> 4) << 3);
    float g = gate[hoff + col];
#pragma unroll
    for (int r = 0; r < 8; ++r) {
      int row = q0 + rb2 + r;
      yg[(rowb + row) * Cc + hoff + col] = (__bf16)(acc[r] * g);
    }
  }
}

// ---------------------------------------------------------------------------
// Host orchestration
// ---------------------------------------------------------------------------
extern "C" void kernel_launch(void* const* d_in, const int* in_sizes, int n_in,
                              void* d_out, int out_size, void* d_ws, size_t ws_size,
                              hipStream_t stream) {
  (void)in_sizes; (void)n_in; (void)out_size; (void)ws_size;
  const int*   tokens    = (const int*)d_in[0];
  const float* tok_emb   = (const float*)d_in[1];
  const float* pos_emb   = (const float*)d_in[2];
  const float* memw      = (const float*)d_in[3];
  const float* ln1_w     = (const float*)d_in[4];
  const float* ln1_b     = (const float*)d_in[5];
  const float* qkv_w     = (const float*)d_in[6];
  const float* qkv_b     = (const float*)d_in[7];
  const float* proj_w    = (const float*)d_in[8];
  const float* proj_b    = (const float*)d_in[9];
  const float* attn_gate = (const float*)d_in[10];
  const float* th1_w     = (const float*)d_in[11];
  const float* th1_b     = (const float*)d_in[12];
  const float* th2_w     = (const float*)d_in[13];
  const float* th2_b     = (const float*)d_in[14];
  const float* ln2_w     = (const float*)d_in[15];
  const float* ln2_b     = (const float*)d_in[16];
  const float* gate_w    = (const float*)d_in[17];
  const float* gate_b    = (const float*)d_in[18];
  const float* e1_w      = (const float*)d_in[19];
  const float* e1_b      = (const float*)d_in[20];
  const float* e2_w      = (const float*)d_in[21];
  const float* e2_b      = (const float*)d_in[22];
  const float* lnf_w     = (const float*)d_in[23];
  const float* lnf_b     = (const float*)d_in[24];
  const float* head_w    = (const float*)d_in[25];
  const float* head_b    = (const float*)d_in[26];

  char* ws = (char*)d_ws;
  size_t off = 0;
  auto take = [&](size_t bytes) -> char* {
    off = (off + 255) & ~(size_t)255;
    char* p = ws + off;
    off += bytes;
    return p;
  };
  float*  x     = (float*)take((size_t)Rr * Cc * 4);
  __bf16* hbuf  = (__bf16*)take((size_t)Rr * Cc * 2);
  __bf16* qkv   = (__bf16*)take((size_t)Rr * C3 * 2);
  __bf16* yg    = (__bf16*)take((size_t)Rr * Cc * 2);
  __bf16* tb    = (__bf16*)take((size_t)Rr * TDc * 2);
  __bf16* he    = (__bf16*)take((size_t)Rr * F4 * 2);
  float*  wgate = (float*)take((size_t)Rr * Ee * 4);
  int*    counts= (int*)take((size_t)Ee * 4);
  int*    lists = (int*)take((size_t)Ee * Rr * 4);
  __bf16* wq    = (__bf16*)take((size_t)Lc * Cc * C3 * 2);   // all weights: W^T (N x K)
  __bf16* wp    = (__bf16*)take((size_t)Lc * Cc * Cc * 2);
  __bf16* wt1   = (__bf16*)take((size_t)Lc * Cc * TDc * 2);
  __bf16* wt2   = (__bf16*)take((size_t)Lc * TDc * Cc * 2);
  __bf16* we1   = (__bf16*)take((size_t)Lc * Ee * Cc * F4 * 2);
  __bf16* we2   = (__bf16*)take((size_t)Lc * Ee * F4 * Cc * 2);
  __bf16* wh    = (__bf16*)take((size_t)Cc * Vv * 2);
  __bf16* wm    = (__bf16*)take((size_t)Mm * Cc * 2);

  // convert + transpose all GEMM weights once (batched over matrices)
  auto f2bfT = [&](const float* src, __bf16* dst, int K, int N, int nmat) {
    dim3 grid((unsigned)(((long)K * N + 255) / 256), nmat);
    k_f2bfT<<<grid, 256, 0, stream>>>(src, dst, K, N);
  };
  f2bfT(qkv_w, wq, Cc, C3, Lc);
  f2bfT(proj_w, wp, Cc, Cc, Lc);
  f2bfT(th1_w, wt1, Cc, TDc, Lc);
  f2bfT(th2_w, wt2, TDc, Cc, Lc);
  f2bfT(e1_w, we1, Cc, F4, Lc * Ee);
  f2bfT(e2_w, we2, F4, Cc, Lc * Ee);
  f2bfT(head_w, wh, Cc, Vv, 1);
  k_f2bf<<<(unsigned)(((size_t)Mm * Cc + 255) / 256), 256, 0, stream>>>(
      memw, wm, (long)Mm * Cc);

  // specialized GEMM launchers (straight-line epilogues)
  auto gemm_plain = [&](const __bf16* A, int lda, const __bf16* Bt, int N, int K,
                        const float* bias, float* outF, __bf16* outB) {
    dim3 grid(N / 128, Rr / 64);
    k_gemm<0, 0, 0><<<grid, 256, 0, stream>>>(A, lda, Bt, N, K, Rr, bias, outF,
                                              outB, nullptr, nullptr, nullptr, 0);
  };
  auto gemm_res = [&](const __bf16* A, int lda, const __bf16* Bt, int N, int K,
                      const float* bias, float* outF, __bf16* outB) {
    dim3 grid(N / 128, Rr / 64);
    k_gemm<0, 1, 0><<<grid, 256, 0, stream>>>(A, lda, Bt, N, K, Rr, bias, outF,
                                              outB, nullptr, nullptr, nullptr, 0);
  };
  auto gemm_tanh = [&](const __bf16* A, int lda, const __bf16* Bt, int N, int K,
                       const float* bias, __bf16* outB) {
    dim3 grid(N / 128, Rr / 64);
    k_gemm<1, 0, 0><<<grid, 256, 0, stream>>>(A, lda, Bt, N, K, Rr, bias, nullptr,
                                              outB, nullptr, nullptr, nullptr, 0);
  };
  auto gemm_gelu_gather = [&](const __bf16* A, int lda, const __bf16* Bt, int N,
                              int K, const float* bias, __bf16* outB,
                              const int* gidx, const int* gcnt) {
    dim3 grid(N / 128, Rr / 64);
    k_gemm<2, 0, 1><<<grid, 256, 0, stream>>>(A, lda, Bt, N, K, Rr, bias, nullptr,
                                              outB, gidx, gcnt, nullptr, 0);
  };
  auto gemm_scatter = [&](const __bf16* A, int lda, const __bf16* Bt, int N, int K,
                          const float* bias, float* outF, const int* gidx,
                          const int* gcnt, const float* wts, int expert) {
    dim3 grid(N / 128, Rr / 64);
    k_gemm<0, 2, 0><<<grid, 256, 0, stream>>>(A, lda, Bt, N, K, Rr, bias, outF,
                                              nullptr, gidx, gcnt, wts, expert);
  };

  k_embed<<<Rr, 128, 0, stream>>>(tokens, tok_emb, pos_emb, x);

  for (int l = 0; l < Lc; ++l) {
    // LN1 -> h (bf16)
    k_layernorm<<<Rr, 128, 0, stream>>>(x, ln1_w + l * Cc, ln1_b + l * Cc, hbuf);
    // qkv = h @ qkv_w + b  (bf16 out only)
    gemm_plain(hbuf, Cc, wq + (size_t)l * Cc * C3, C3, Cc, qkv_b + l * C3,
               nullptr, qkv);
    // fused attention -> yg = (attn output) * attn_gate (bf16)
    {
      dim3 grid(Tt / 32, Hh, Bb);
      k_attn<<<grid, 128, 0, stream>>>(qkv, wm, attn_gate + l * Cc, yg);
    }
    // x += yg @ proj_w + proj_b ; h <- bf16(x)
    gemm_res(yg, Cc, wp + (size_t)l * Cc * Cc, Cc, Cc, proj_b + l * Cc, x, hbuf);
    // tb = tanh(x @ th1 + b)
    gemm_tanh(hbuf, Cc, wt1 + (size_t)l * Cc * TDc, TDc, Cc, th1_b + l * TDc, tb);
    // x += tb @ th2 + b
    gemm_res(tb, TDc, wt2 + (size_t)l * TDc * Cc, Cc, TDc, th2_b + l * Cc,
             x, nullptr);
    // LN2 -> h2 (bf16)
    k_layernorm<<<Rr, 128, 0, stream>>>(x, ln2_w + l * Cc, ln2_b + l * Cc, hbuf);
    // gate + top-2 + expert row lists
    k_zero_int<<<1, 32, 0, stream>>>(counts, Ee);
    k_gate<<<Rr / 8, 256, 0, stream>>>(hbuf, gate_w + (size_t)l * Cc * Ee,
                                       gate_b + l * Ee, wgate, counts, lists);
    // experts: gathered rows only (top-2 => ~half of dense reference work)
    for (int e = 0; e < Ee; ++e) {
      const int* gidx = lists + (size_t)e * Rr;
      const int* gcnt = counts + e;
      // he = gelu(h2[gather] @ e1 + b1)  (compacted rows, bf16)
      gemm_gelu_gather(hbuf, Cc, we1 + (size_t)(l * Ee + e) * Cc * F4, F4, Cc,
                       e1_b + (size_t)(l * Ee + e) * F4, he, gidx, gcnt);
      // x[gidx] += w * (he @ e2 + b2)
      gemm_scatter(he, F4, we2 + (size_t)(l * Ee + e) * F4 * Cc, Cc, F4,
                   e2_b + (size_t)(l * Ee + e) * Cc, x, gidx, gcnt, wgate, e);
    }
  }

  // final LN + head
  k_layernorm<<<Rr, 128, 0, stream>>>(x, lnf_w, lnf_b, hbuf);
  gemm_plain(hbuf, Cc, wh, Vv, Cc, head_b, (float*)d_out, nullptr);
}
